// TreeAttention_14087492731509
// MI455X (gfx1250) — matrix operand order repaired and checked by reading, not compiled, not run on hardware
//
#include <hip/hip_runtime.h>
#include <hip/hip_bf16.h>

// MI455X / gfx1250 attention layer: bf16 WMMA + TDM double-buffered tiles.
// B=2, S=2048, D=2048, H=16, HD=128.

typedef __bf16 bf16;
typedef __attribute__((ext_vector_type(8)))  __bf16 bf16x8;
typedef __attribute__((ext_vector_type(16))) __bf16 bf16x16;
typedef __attribute__((ext_vector_type(8)))  float  v8f;
typedef __attribute__((ext_vector_type(4)))  unsigned int u32x4;
typedef __attribute__((ext_vector_type(8)))  int    i32x8;
typedef __attribute__((ext_vector_type(4)))  int    i32x4;

static constexpr int Bc  = 2;
static constexpr int Sc  = 2048;
static constexpr int Dc  = 2048;
static constexpr int Hc  = 16;
static constexpr int HDc = 128;

#define DEVINL static __device__ __forceinline__

DEVINL bf16x16 ld_frag(const bf16* p0, const bf16* p1) {
  bf16x8 lo = *(const bf16x8*)p0;
  bf16x8 hi = *(const bf16x8*)p1;
  bf16x16 r;
#pragma unroll
  for (int i = 0; i < 8; ++i) { r[i] = lo[i]; r[i + 8] = hi[i]; }
  return r;
}

DEVINL v8f wmma_bf16(bf16x16 a, bf16x16 b, v8f c) {
  return __builtin_amdgcn_wmma_f32_16x16x32_bf16(false, a, false, b,
                                                 (short)0, c, false, false);
}

DEVINL v8f zero8() {
  v8f z;
#pragma unroll
  for (int i = 0; i < 8; ++i) z[i] = 0.0f;
  return z;
}

DEVINL unsigned lds_off(const void* p) {
  // generic LDS pointer = aperture(hi32) | byte-offset(lo32)
  return (unsigned)(unsigned long long)(size_t)p;
}

// ---------------------------------------------------------------------------
// TDM: async 2-D tile load global -> LDS (bf16 elements, data_size=2B).
// tile_d0 = contiguous elems/row, tile_d1 = rows, stride0 in elements.
// pad codes: LDS row padding so rows land on bank-conflict-free strides.
//   interval code i -> pad every 2^(i+1) DWORDs; amount code a -> (a+1) DWORDs.
// ---------------------------------------------------------------------------
DEVINL void tdm_load_2d(const void* gsrc, unsigned lds_byte,
                        unsigned tensor_d0, unsigned tensor_d1,
                        unsigned tile_d0, unsigned tile_d1,
                        unsigned long long stride0,
                        unsigned pad_int_code, unsigned pad_amt_code) {
  unsigned long long ga = (unsigned long long)(size_t)gsrc;
  u32x4 g0;
  g0[0] = 1u;                                               // count=1 (valid)
  g0[1] = lds_byte;                                         // lds_addr
  g0[2] = (unsigned)ga;                                     // global_addr lo
  g0[3] = (unsigned)((ga >> 32) & 0x01FFFFFFull) | (2u << 30); // addr hi | type=2
  i32x8 g1;
  g1[0] = (int)((1u << 16)                                  // data_size = 2B
              | (1u << 20)                                  // pad_enable
              | (pad_int_code << 22) | (pad_amt_code << 25));
  g1[1] = (int)((tensor_d0 & 0xFFFFu) << 16);               // td0[15:0]
  g1[2] = (int)((tensor_d0 >> 16) | ((tensor_d1 & 0xFFFFu) << 16));
  g1[3] = (int)((tensor_d1 >> 16) | (tile_d0 << 16));       // td1 hi | tile_d0
  g1[4] = (int)(tile_d1);                                   // tile_d1 (tile_d2=0)
  g1[5] = (int)(unsigned)(stride0 & 0xFFFFFFFFull);         // dim0_stride lo
  g1[6] = (int)(unsigned)((stride0 >> 32) & 0xFFFFull);     // dim0_stride hi
  g1[7] = 0;                                                // dim1_stride (unused, 2D)
  i32x4 z4; z4[0] = 0; z4[1] = 0; z4[2] = 0; z4[3] = 0;
  i32x8 z8;
#pragma unroll
  for (int i = 0; i < 8; ++i) z8[i] = 0;
  // 6-arg (clang-23 / therock) form: (g0, g1, g2, g3, g4, cpol)
  __builtin_amdgcn_tensor_load_to_lds(g0, g1, z4, z4, z8, 0);
}

// ---------------------------------------------------------------------------
// fp32 -> bf16 conversion
// ---------------------------------------------------------------------------
__global__ __launch_bounds__(256) void cvt_bf16_kernel(const float* __restrict__ src,
                                                       bf16* __restrict__ dst, int n) {
  int i = blockIdx.x * 256 + threadIdx.x;
  if (i < n) dst[i] = (bf16)src[i];
}

// ---------------------------------------------------------------------------
// GEMM: C[M x N] = A[M x K] * W[N x K]^T  (bf16 in, fp32 acc)
// Block 256 thr = 8 waves (4x2); block tile 256x128; wave tile 64x64 (4x4 WMMA).
// K-tiles of 32 streamed by TDM, double-buffered.
// MODE 0: bf16 out. MODE 1: bf16 out transposed to Vt(B, D, S). MODE 2: f32 out.
// ---------------------------------------------------------------------------
static constexpr int LDA = 40; // 32 K-elems + 8 pad (80B rows, conflict-free)

template <int MODE>
__global__ __launch_bounds__(256) void gemm_wmma_kernel(const bf16* __restrict__ A,
                                                        const bf16* __restrict__ W,
                                                        void* __restrict__ Cout) {
  constexpr int K = 2048, N = 2048;
  __shared__ __align__(16) bf16 sA[2][256 * LDA];   // 2 x 20 KiB
  __shared__ __align__(16) bf16 sB[2][128 * LDA];   // 2 x 10 KiB

  const int tid  = threadIdx.x;
  const int lane = tid & 31;
  const int wave = tid >> 5;
  const int wm   = wave >> 1;          // 0..3
  const int wn   = wave & 1;           // 0..1
  const int bm   = blockIdx.y * 256;
  const int bn   = blockIdx.x * 128;
  const int cg   = lane & 15;
  const int kh   = lane >> 4;

  v8f acc[4][4];
#pragma unroll
  for (int i = 0; i < 4; ++i)
#pragma unroll
    for (int j = 0; j < 4; ++j) acc[i][j] = zero8();

  // TDM stage issue: A tile 256 rows x 32, B tile 128 rows x 32
  auto issue = [&](int buf, int k0) {
    tdm_load_2d(&A[(size_t)bm * K + k0], lds_off(&sA[buf][0]),
                K, 4096, 32, 256, K, 3, 3);
    tdm_load_2d(&W[(size_t)bn * K + k0], lds_off(&sB[buf][0]),
                K, 2048, 32, 128, K, 3, 3);
  };

  if (wave == 0) issue(0, 0);
  int buf = 0;
  for (int k0 = 0; k0 < K; k0 += 32) {
    if (wave == 0) {
      if (k0 + 32 < K) {
        issue(buf ^ 1, k0 + 32);
        __builtin_amdgcn_s_wait_tensorcnt(2);   // stage k0 retired (in-order)
      } else {
        __builtin_amdgcn_s_wait_tensorcnt(0);
      }
    }
    __syncthreads();

    bf16x16 afr[4], bfr[4];
#pragma unroll
    for (int i = 0; i < 4; ++i) {
      const bf16* base = &sA[buf][(wm * 64 + i * 16 + cg) * LDA];
      afr[i] = ld_frag(base + kh * 8, base + 16 + kh * 8);
    }
#pragma unroll
    for (int j = 0; j < 4; ++j) {
      const bf16* base = &sB[buf][(wn * 64 + j * 16 + cg) * LDA];
      bfr[j] = ld_frag(base + kh * 8, base + 16 + kh * 8);
    }
#pragma unroll
    for (int i = 0; i < 4; ++i)
#pragma unroll
      for (int j = 0; j < 4; ++j) acc[i][j] = wmma_bf16(afr[i], bfr[j], acc[i][j]);
    __syncthreads();
    buf ^= 1;
  }

  const int rs = lane >> 4;
#pragma unroll
  for (int i = 0; i < 4; ++i) {
#pragma unroll
    for (int j = 0; j < 4; ++j) {
      int col = bn + wn * 64 + j * 16 + cg;
#pragma unroll
      for (int v = 0; v < 8; ++v) {
        int   row = bm + wm * 64 + i * 16 + v + 8 * rs;
        float val = acc[i][j][v];
        if (MODE == 0) {
          ((bf16*)Cout)[(size_t)row * N + col] = (bf16)val;
        } else if (MODE == 1) {
          int bb = row >> 11, s = row & (Sc - 1);
          ((bf16*)Cout)[((size_t)(bb * Dc + col)) * Sc + s] = (bf16)val;
        } else {
          ((float*)Cout)[(size_t)row * N + col] = val;
        }
      }
    }
  }
}

// ---------------------------------------------------------------------------
// RoPE in-place on Q and K (B,S,H,HD). One thread per rotation pair.
// ---------------------------------------------------------------------------
__global__ __launch_bounds__(256) void rope_kernel(bf16* __restrict__ q,
                                                   bf16* __restrict__ k,
                                                   const float* __restrict__ pos) {
  int idx = blockIdx.x * 256 + threadIdx.x;        // B*S*H*64
  int j = idx & 63;
  int h = (idx >> 6) & (Hc - 1);
  int s = (idx >> 10) & (Sc - 1);
  int b = idx >> 21;
  float p     = pos[((size_t)(b * Sc + s)) * 4 + (j & 3)];
  float theta = __powf(10000.0f, -(float)(j >> 2) / 16.0f);
  float sn, cs;
  __sincosf(p * theta, &sn, &cs);
  size_t base = (((size_t)(b * Sc + s) * Hc + h) * HDc) + 2 * j;
  float qr = (float)q[base], qi = (float)q[base + 1];
  q[base]     = (bf16)(qr * cs - qi * sn);
  q[base + 1] = (bf16)(qr * sn + qi * cs);
  float kr = (float)k[base], ki = (float)k[base + 1];
  k[base]     = (bf16)(kr * cs - ki * sn);
  k[base + 1] = (bf16)(kr * sn + ki * cs);
}

// ---------------------------------------------------------------------------
// Flash attention. Grid (S/128, H, B), block 256 (8 waves x 16 queries).
// K/V tiles streamed by TDM, double-buffered. 16 WMMAs per 32-key block/wave.
// ---------------------------------------------------------------------------
static constexpr int LDK = 136; // 128 + 8 pad
static constexpr int LDP = 40;  // 32 + 8 pad

__global__ __launch_bounds__(256) void flash_attn_kernel(const bf16* __restrict__ Q,
                                                         const bf16* __restrict__ Km,
                                                         const bf16* __restrict__ Vt,
                                                         bf16* __restrict__ Ctx) {
  __shared__ __align__(16) bf16 sK[2][32 * LDK];    // [key][hd]
  __shared__ __align__(16) bf16 sV[2][128 * LDP];   // [hd][key]
  __shared__ __align__(16) bf16 sP[8][16 * LDP];    // per-wave P staging

  const int tid  = threadIdx.x;
  const int lane = tid & 31;
  const int wave = tid >> 5;
  const int b    = blockIdx.z;
  const int h    = blockIdx.y;
  const int q0   = blockIdx.x * 128;
  const int qw   = q0 + wave * 16;
  const int cg   = lane & 15;
  const int hi   = lane >> 4;

  bf16x16 qfr[4];
  {
    const bf16* qrow = Q + ((size_t)(b * Sc + qw + cg) * Hc + h) * HDc;
#pragma unroll
    for (int ks = 0; ks < 4; ++ks)
      qfr[ks] = ld_frag(qrow + ks * 32 + hi * 8, qrow + ks * 32 + 16 + hi * 8);
  }

  v8f   O[8];
  float mrow[8], lrow[8];
#pragma unroll
  for (int t = 0; t < 8; ++t) O[t] = zero8();
#pragma unroll
  for (int v = 0; v < 8; ++v) { mrow[v] = -1e30f; lrow[v] = 0.0f; }

  const float scale = 0.08838834764831845f; // 1/sqrt(128)
  const int   nkb   = (q0 + 128) >> 5;

  auto issue = [&](int bufi, int k0) {
    // K tile: 32 keys x 128 hd, row stride H*HD; LDS rows padded to 136 elems
    tdm_load_2d(&Km[((size_t)(b * Sc + k0) * Hc + h) * HDc], lds_off(&sK[bufi][0]),
                (unsigned)(Hc * HDc), Sc, 128, 32,
                (unsigned long long)(Hc * HDc), 5, 3);
    // Vt slice: 128 hd x 32 keys, row stride S; LDS rows padded to 40 elems
    tdm_load_2d(&Vt[((size_t)(b * Hc + h) * HDc) * Sc + k0], lds_off(&sV[bufi][0]),
                Sc, HDc, 32, 128, Sc, 3, 3);
  };

  if (wave == 0) issue(0, 0);
  int buf = 0;
  for (int kb = 0; kb < nkb; ++kb) {
    const int k0 = kb * 32;
    if (wave == 0) {
      if (kb + 1 < nkb) {
        issue(buf ^ 1, k0 + 32);
        __builtin_amdgcn_s_wait_tensorcnt(2);
      } else {
        __builtin_amdgcn_s_wait_tensorcnt(0);
      }
    }
    __syncthreads();

    // scores: 2 key-tiles of 16, K-dim = HD in 4 steps of 32
    v8f sc[2];
    sc[0] = zero8(); sc[1] = zero8();
#pragma unroll
    for (int ks = 0; ks < 4; ++ks) {
#pragma unroll
      for (int nt = 0; nt < 2; ++nt) {
        const bf16* base = &sK[buf][(nt * 16 + cg) * LDK + ks * 32 + hi * 8];
        sc[nt] = wmma_bf16(qfr[ks], ld_frag(base, base + 16), sc[nt]);
      }
    }

    // online softmax (rows: m = v + 8*hi; cols: key = k0 + nt*16 + cg)
#pragma unroll
    for (int v = 0; v < 8; ++v) {
      const int row = qw + v + 8 * hi;
      float s0 = sc[0][v] * scale + ((k0 + cg)      <= row ? 0.0f : -1e9f);
      float s1 = sc[1][v] * scale + ((k0 + 16 + cg) <= row ? 0.0f : -1e9f);
      float t  = fmaxf(s0, s1);
#pragma unroll
      for (int off = 1; off < 16; off <<= 1) t = fmaxf(t, __shfl_xor(t, off, 32));
      const float mnew  = fmaxf(mrow[v], t);
      const float alpha = __expf(mrow[v] - mnew);
      mrow[v] = mnew;
      float p0 = __expf(s0 - mnew);
      float p1 = __expf(s1 - mnew);
      float rs = p0 + p1;
#pragma unroll
      for (int off = 1; off < 16; off <<= 1) rs += __shfl_xor(rs, off, 32);
      lrow[v] = lrow[v] * alpha + rs;
#pragma unroll
      for (int t8 = 0; t8 < 8; ++t8) O[t8][v] *= alpha;
      sP[wave][(v + 8 * hi) * LDP + cg]      = (bf16)p0;
      sP[wave][(v + 8 * hi) * LDP + 16 + cg] = (bf16)p1;
    }
    asm volatile("s_wait_dscnt 0" ::: "memory");  // in-wave LDS store->load

    // P @ V
    {
      const bf16* pb  = &sP[wave][cg * LDP];
      bf16x16     pfr = ld_frag(pb + hi * 8, pb + 16 + hi * 8);
#pragma unroll
      for (int t = 0; t < 8; ++t) {
        const bf16* vb = &sV[buf][(t * 16 + cg) * LDP + hi * 8];
        O[t] = wmma_bf16(pfr, ld_frag(vb, vb + 16), O[t]);
      }
    }
    __syncthreads();
    buf ^= 1;
  }

#pragma unroll
  for (int v = 0; v < 8; ++v) {
    const float inv = 1.0f / lrow[v];
    const int   row = qw + v + 8 * hi;
#pragma unroll
    for (int t = 0; t < 8; ++t)
      Ctx[((size_t)(b * Sc + row) * Hc + h) * HDc + t * 16 + cg] =
          (bf16)(O[t][v] * inv);
  }
}

// ---------------------------------------------------------------------------
// host launcher
// ---------------------------------------------------------------------------
extern "C" void kernel_launch(void* const* d_in, const int* in_sizes, int n_in,
                              void* d_out, int out_size, void* d_ws, size_t ws_size,
                              hipStream_t stream) {
  (void)in_sizes; (void)n_in; (void)out_size; (void)ws_size;
  const float* x   = (const float*)d_in[0];
  const float* pos = (const float*)d_in[1];
  // d_in[2] = mask: causal(-1e9), applied analytically (identical values)
  const float* wq  = (const float*)d_in[3];
  const float* wk  = (const float*)d_in[4];
  const float* wv  = (const float*)d_in[5];
  const float* wo  = (const float*)d_in[6];

  constexpr size_t MiB = 1u << 20;
  char* ws = (char*)d_ws;
  bf16* xb   = (bf16*)(ws + 0 * MiB);
  bf16* wqb  = (bf16*)(ws + 16 * MiB);
  bf16* wkb  = (bf16*)(ws + 24 * MiB);
  bf16* wvb  = (bf16*)(ws + 32 * MiB);
  bf16* wob  = (bf16*)(ws + 40 * MiB);
  bf16* qb   = (bf16*)(ws + 48 * MiB);
  bf16* kbuf = (bf16*)(ws + 64 * MiB);
  bf16* vt   = (bf16*)(ws + 80 * MiB);
  bf16* ctx  = (bf16*)(ws + 96 * MiB);   // total 112 MiB

  const int nX = Bc * Sc * Dc;
  const int nW = Dc * Dc;
  cvt_bf16_kernel<<<(nX + 255) / 256, 256, 0, stream>>>(x,  xb,  nX);
  cvt_bf16_kernel<<<(nW + 255) / 256, 256, 0, stream>>>(wq, wqb, nW);
  cvt_bf16_kernel<<<(nW + 255) / 256, 256, 0, stream>>>(wk, wkb, nW);
  cvt_bf16_kernel<<<(nW + 255) / 256, 256, 0, stream>>>(wv, wvb, nW);
  cvt_bf16_kernel<<<(nW + 255) / 256, 256, 0, stream>>>(wo, wob, nW);

  dim3 ggrid(Dc / 128, (Bc * Sc) / 256);  // (16, 16)
  gemm_wmma_kernel<0><<<ggrid, 256, 0, stream>>>(xb, wqb, qb);
  gemm_wmma_kernel<0><<<ggrid, 256, 0, stream>>>(xb, wkb, kbuf);
  gemm_wmma_kernel<1><<<ggrid, 256, 0, stream>>>(xb, wvb, vt);

  const int nR = Bc * Sc * Hc * (HDc / 2);
  rope_kernel<<<(nR + 255) / 256, 256, 0, stream>>>(qb, kbuf, pos);

  flash_attn_kernel<<<dim3(Sc / 128, Hc, Bc), 256, 0, stream>>>(qb, kbuf, vt, ctx);

  gemm_wmma_kernel<2><<<ggrid, 256, 0, stream>>>(ctx, wob, d_out);
}
